// MultiRNNModel_58291296141913
// MI455X (gfx1250) — matrix operand order repaired and checked by reading, not compiled, not run on hardware
//
#include <hip/hip_runtime.h>

typedef __attribute__((ext_vector_type(16))) _Float16 v16h;
typedef __attribute__((ext_vector_type(8)))  float    v8f;
typedef __attribute__((ext_vector_type(4)))  unsigned v4u;
typedef __attribute__((ext_vector_type(8)))  int      v8i_;
typedef __attribute__((ext_vector_type(4)))  int      v4i_;

#define BATCH   (1 << 20)
#define AST     68            // k-stride (in halves) for LDS A/B tiles: 136B, bank-friendly
#define WAVES   4             // waves per block (128 threads)
#define TPW     4             // batch tiles per wave
#define NTILES_TOTAL (BATCH / 16)
#define NBLOCKS (NTILES_TOTAL / (WAVES * TPW))   // 4096

#if defined(__has_builtin)
#if __has_builtin(__builtin_amdgcn_tensor_load_to_lds) && \
    __has_builtin(__builtin_amdgcn_s_wait_tensorcnt)
#define HAVE_TDM 1
#endif
#endif

__device__ __forceinline__ float sigmoidf_(float x) { return 1.0f / (1.0f + __expf(-x)); }

// Per-wave LDS buffers only: LDS ops from one wave complete in order (ISA 7.1),
// so cross-lane store->load inside a wave needs only a compiler scheduling fence.
__device__ __forceinline__ void wave_sync() { __builtin_amdgcn_wave_barrier(); }

__device__ __forceinline__ v8f wmma_f16(v16h a, v16h b, v8f c) {
  // D = A(16x32 f16) * B(32x16 f16) + C(16x16 f32)
  return __builtin_amdgcn_wmma_f32_16x16x32_f16(false, a, false, b, (short)0, c, false, false);
}

#ifdef HAVE_TDM
// TDM: async DMA of 544 dwords (one 16-row x 34-col f32 tile) global -> LDS.
// D# per ISA 08 §8.3/8.4: group0 = {count=1, lds_addr, global_addr(57b), type=2},
// group1 = {data_size=4B, tensor_dim0=544, tensor_dim1=1, tile_dim0=544, tile_dim1=1}.
// This toolchain exposes the 6-arg builtin: (v4u, v8i, v4i, v4i, v8i, cpol).
__device__ __forceinline__ void tdm_load_tile(const float* gsrc, float* ldst) {
  unsigned lds = (unsigned)(unsigned long long)ldst;          // low 32 bits = LDS offset
  unsigned long long ga = (unsigned long long)gsrc;
  v4u g0;
  g0[0] = 1u;                                                 // count=1 (valid), no gather
  g0[1] = lds;                                                // lds_addr (bytes)
  g0[2] = (unsigned)ga;                                       // global_addr[31:0]
  g0[3] = (unsigned)((ga >> 32) & 0x01FFFFFFu) | 0x80000000u; // addr[56:32] | type=2
  v8i_ g1;
  g1[0] = (int)(2u << 16);    // data_size=2 (4 bytes); mask/flags/pad = 0
  g1[1] = (int)(544u << 16);  // tensor_dim0[15:0]=544 in bits[31:16]
  g1[2] = (int)(1u << 16);    // tensor_dim0 hi=0; tensor_dim1[15:0]=1
  g1[3] = (int)(544u << 16);  // tensor_dim1 hi=0; tile_dim0=544
  g1[4] = 1;                  // tile_dim1=1, tile_dim2=0
  g1[5] = 544;                // tensor_dim0_stride[31:0]
  g1[6] = 0;                  // stride hi / tensor_dim1_stride lo
  g1[7] = 0;
  v4i_ z4 = {0, 0, 0, 0};
  v8i_ z8 = {0, 0, 0, 0, 0, 0, 0, 0};
  __builtin_amdgcn_tensor_load_to_lds(g0, g1, z4, z4, z8, 0);
}
#endif

// A-matrix 16x32 f16 fragment (ISA 7.12.2): lane m = lane&15, half = lane>>4.
__device__ __forceinline__ v16h load_a_frag(const _Float16* A, int m, int half, int k0) {
  union { v16h v; unsigned u[8]; } r;
  const unsigned* base = (const unsigned*)(A + m * AST);
  #pragma unroll
  for (int i = 0; i < 8; ++i) {
    int kk = k0 + ((i & 3) << 1) + ((i >> 2) << 4) + (half << 3);
    r.u[i] = base[kk >> 1];
  }
  return r.v;
}

// B-matrix 32x16 f16 fragment: lane n = lane&15; VGPR v: K = 2(v&3) + 16(v>>2) (+8 for lanes 16..31).
__device__ __forceinline__ v16h load_b_frag(const _Float16* Bt, int n, int half, int k0) {
  union { v16h v; unsigned u[8]; } r;
  const unsigned* base = (const unsigned*)(Bt + n * AST);
  #pragma unroll
  for (int i = 0; i < 8; ++i) {
    int kk = k0 + ((i & 3) << 1) + ((i >> 2) << 4) + (half << 3);
    r.u[i] = base[kk >> 1];
  }
  return r.v;
}

// One FC layer on a 16-row tile: Adst = relu(Asrc * Bt + bias), zero-padded to 64 cols.
template <int KSTEPS, int NTILES>
__device__ __forceinline__ void fc_relu(const _Float16* Asrc, const _Float16* Bt,
                                        const float* bias, _Float16* Adst,
                                        int l15, int half, int realN) {
  v16h a[KSTEPS];
  #pragma unroll
  for (int s = 0; s < KSTEPS; ++s) a[s] = load_a_frag(Asrc, l15, half, 32 * s);
  #pragma unroll
  for (int t = 0; t < 4; ++t) {
    int col = t * 16 + l15;
    if (t < NTILES) {
      v8f acc = {};
      #pragma unroll
      for (int s = 0; s < KSTEPS; ++s) {
        v16h b = load_b_frag(Bt, col, half, 32 * s);
        acc = wmma_f16(a[s], b, acc);
      }
      float bv = bias[col];
      #pragma unroll
      for (int v = 0; v < 8; ++v) {
        int row = half * 8 + v;                    // D layout: M = v + 8*half, N = lane&15
        float val = (col < realN) ? fmaxf(acc[v] + bv, 0.0f) : 0.0f;
        Adst[row * AST + col] = (_Float16)val;
      }
    } else {
      #pragma unroll
      for (int v = 0; v < 8; ++v) Adst[(half * 8 + v) * AST + col] = (_Float16)0.0f;
    }
  }
}

__global__ __launch_bounds__(128) void multirnn_fused_wmma_kernel(
    const float* __restrict__ in,  const float* __restrict__ Wih,
    const float* __restrict__ bih, const float* __restrict__ bhh,
    const float* __restrict__ W1,  const float* __restrict__ b1,
    const float* __restrict__ W2,  const float* __restrict__ b2,
    const float* __restrict__ W3,  const float* __restrict__ b3,
    const float* __restrict__ W4,  const float* __restrict__ b4,
    float* __restrict__ out) {
  // ---- LDS ----
  __shared__ _Float16 sW1[64 * AST];   // Bt layout: [n][k], k-stride AST
  __shared__ _Float16 sW2[64 * AST];
  __shared__ _Float16 sW3[48 * AST];
  __shared__ _Float16 sW4[16 * AST];
  __shared__ float sB1[64], sB2[64], sB3[64], sB4[16];
  __shared__ float sWih[176], sBg[88];
  __shared__ _Float16 sA[WAVES][2][16 * AST];               // ping-pong activation tiles
  __shared__ __align__(16) float sRaw[WAVES][2][544];       // double-buffered input/output staging

  const int tid  = threadIdx.x;
  const int wave = tid >> 5;
  const int lane = tid & 31;
  const int l15  = lane & 15;
  const int half = lane >> 4;

  const int tile0 = (blockIdx.x * WAVES + wave) * TPW;      // first 16-row tile of this wave

#ifdef HAVE_TDM
  // Kick off the first tile's DMA immediately; it overlaps the weight staging below.
  tdm_load_tile(in + (size_t)tile0 * 16 * 34, &sRaw[wave][0][0]);
#endif

  // ---- one-time per-block weight staging (f16, zero-padded) ----
  for (int i = tid; i < 64 * AST; i += 128) { sW1[i] = (_Float16)0.0f; sW2[i] = (_Float16)0.0f; }
  for (int i = tid; i < 48 * AST; i += 128) sW3[i] = (_Float16)0.0f;
  for (int i = tid; i < 16 * AST; i += 128) sW4[i] = (_Float16)0.0f;
  __syncthreads();
  for (int i = tid; i < 51 * 34; i += 128) sW1[(i / 34) * AST + (i % 34)] = (_Float16)W1[i];
  for (int i = tid; i < 51 * 51; i += 128) sW2[(i / 51) * AST + (i % 51)] = (_Float16)W2[i];
  for (int i = tid; i < 34 * 51; i += 128) sW3[(i / 51) * AST + (i % 51)] = (_Float16)W3[i];
  for (int i = tid; i < 5 * 34;  i += 128) sW4[(i / 34) * AST + (i % 34)] = (_Float16)W4[i];
  if (tid < 64) {
    sB1[tid] = (tid < 51) ? b1[tid] : 0.0f;
    sB2[tid] = (tid < 51) ? b2[tid] : 0.0f;
    sB3[tid] = (tid < 34) ? b3[tid] : 0.0f;
  }
  if (tid < 16) sB4[tid] = (tid < 5) ? b4[tid] : 0.0f;
  for (int i = tid; i < 176; i += 128) sWih[i] = Wih[i];
  for (int i = tid; i < 88;  i += 128) sBg[i] = bih[i] + bhh[i];
  __syncthreads();

  _Float16* A0 = &sA[wave][0][0];
  _Float16* A1 = &sA[wave][1][0];

  for (int it = 0; it < TPW; ++it) {
    const int tile = tile0 + it;
    const size_t row0 = (size_t)tile * 16;
    float* raw = &sRaw[wave][it & 1][0];

#ifdef HAVE_TDM
    // Prefetch next tile into the other buffer, then wait for the current one.
    if (it + 1 < TPW) {
      tdm_load_tile(in + (row0 + 16) * 34, &sRaw[wave][(it + 1) & 1][0]);
      __builtin_amdgcn_s_wait_tensorcnt(1);   // in-order: older (current) transfer done
    } else {
      __builtin_amdgcn_s_wait_tensorcnt(0);
    }
#else
    // Fallback: coalesced vector loads, 16 rows x 34 f32 = 544 floats.
    {
      const float4* src = (const float4*)(in + row0 * 34);
      float4* dv = (float4*)raw;
      #pragma unroll
      for (int j = 0; j < 4; ++j) dv[lane + 32 * j] = src[lane + 32 * j];
      raw[512 + lane] = in[row0 * 34 + 512 + lane];
      if (it + 1 < TPW) __builtin_prefetch(in + (row0 + 16) * 34, 0, 1);
    }
#endif
    wave_sync();

    // ---- LSTM heads + concat others -> A0 (f16, cols 0..33 real, 34..63 zero) ----
    {
      const int r = l15;
      const float* xr = raw + r * 34;
      const int kb = half ? 6 : 0, ke = half ? 11 : 6;
      for (int k = kb; k < ke; ++k) {
        float x0 = xr[2 * k], x1 = xr[2 * k + 1];
        const float* w = sWih + k * 16;   // W_ih[k][g][h], g-major pairs
        #pragma unroll
        for (int j = 0; j < 2; ++j) {
          float gi = fmaf(w[(0 + j) * 2 + 1], x1, fmaf(w[(0 + j) * 2], x0, sBg[k * 8 + 0 + j]));
          float gg = fmaf(w[(4 + j) * 2 + 1], x1, fmaf(w[(4 + j) * 2], x0, sBg[k * 8 + 4 + j]));
          float go = fmaf(w[(6 + j) * 2 + 1], x1, fmaf(w[(6 + j) * 2], x0, sBg[k * 8 + 6 + j]));
          float cc = sigmoidf_(gi) * tanhf(gg);
          float hh = sigmoidf_(go) * tanhf(cc);
          A0[r * AST + 2 * k + j] = (_Float16)hh;
        }
      }
      #pragma unroll
      for (int j = 0; j < 6; ++j) { int c = 22 + half * 6 + j; A0[r * AST + c] = (_Float16)xr[c]; }
      #pragma unroll
      for (int j = 0; j < 15; ++j) { int c = 34 + half * 15 + j; A0[r * AST + c] = (_Float16)0.0f; }
    }
    wave_sync();

    // ---- MLP via WMMA (K padded to 64 => 2 k-steps per layer) ----
    fc_relu<2, 4>(A0, sW1, sB1, A1, l15, half, 51);   // 34 -> 51
    wave_sync();
    fc_relu<2, 4>(A1, sW2, sB2, A0, l15, half, 51);   // 51 -> 51
    wave_sync();
    fc_relu<2, 3>(A0, sW3, sB3, A1, l15, half, 34);   // 51 -> 34
    wave_sync();

    // ---- final layer 34 -> 5, sigmoid, staged via LDS for coalesced store ----
    {
      v16h a0 = load_a_frag(A1, l15, half, 0);
      v16h a1 = load_a_frag(A1, l15, half, 32);
      v8f acc = {};
      acc = wmma_f16(a0, load_b_frag(sW4, l15, half, 0),  acc);
      acc = wmma_f16(a1, load_b_frag(sW4, l15, half, 32), acc);
      if (l15 < 5) {
        float bv = sB4[l15];
        #pragma unroll
        for (int v = 0; v < 8; ++v)
          raw[(half * 8 + v) * 5 + l15] = sigmoidf_(acc[v] + bv);
      }
    }
    wave_sync();
    for (int u = lane; u < 80; u += 32) out[row0 * 5 + u] = raw[u];
    wave_sync();   // protect raw/A buffers before next tile
  }
}

extern "C" void kernel_launch(void* const* d_in, const int* in_sizes, int n_in,
                              void* d_out, int out_size, void* d_ws, size_t ws_size,
                              hipStream_t stream) {
  const float* in  = (const float*)d_in[0];
  const float* Wih = (const float*)d_in[1];
  const float* bih = (const float*)d_in[2];
  const float* bhh = (const float*)d_in[3];
  const float* W1  = (const float*)d_in[4];
  const float* b1  = (const float*)d_in[5];
  const float* W2  = (const float*)d_in[6];
  const float* b2  = (const float*)d_in[7];
  const float* W3  = (const float*)d_in[8];
  const float* b3  = (const float*)d_in[9];
  const float* W4  = (const float*)d_in[10];
  const float* b4  = (const float*)d_in[11];
  float* out = (float*)d_out;
  (void)in_sizes; (void)n_in; (void)out_size; (void)d_ws; (void)ws_size;

  multirnn_fused_wmma_kernel<<<NBLOCKS, 128, 0, stream>>>(
      in, Wih, bih, bhh, W1, b1, W2, b2, W3, b3, W4, b4, out);
}
